// qLSTM_5660766896666
// MI455X (gfx1250) — matrix operand order repaired and checked by reading, not compiled, not run on hardware
//
#include <hip/hip_runtime.h>

// ---------------- problem constants ----------------
constexpr int Bc = 4;
constexpr int Tc = 2048;
constexpr int Dc = 1024;
constexpr int Hc = 2048;      // D * EXP
constexpr int Kc = 4;         // conv taps
constexpr int MT = Bc * Tc;   // 8192 rows for all GEMMs

typedef unsigned short u16;
typedef __attribute__((ext_vector_type(16))) __bf16 v16bf;
typedef __attribute__((ext_vector_type(8)))  __bf16 v8bf;
typedef __attribute__((ext_vector_type(8)))  float  v8f;
typedef __attribute__((ext_vector_type(4)))  unsigned int u32x4;
typedef __attribute__((ext_vector_type(8)))  unsigned int u32x8;

__device__ __forceinline__ u16 f2bf(float f) {
    unsigned int u = __float_as_uint(f);
    u += 0x7FFFu + ((u >> 16) & 1u);   // round-to-nearest-even
    return (u16)(u >> 16);
}

// ---------------- f32 -> bf16 conversion ----------------
__global__ __launch_bounds__(256)
void k_cvt_bf16(const float* __restrict__ src, u16* __restrict__ dst, long n) {
    long i = (long)blockIdx.x * 256 + threadIdx.x;
    long stride = (long)gridDim.x * 256;
    for (; i < n; i += stride) dst[i] = f2bf(src[i]);
}

// ---------------- depthwise causal conv over time ----------------
__global__ __launch_bounds__(256)
void k_conv(const float* __restrict__ gi, const float* __restrict__ Wc,
            const float* __restrict__ bc, float* __restrict__ vf,
            u16* __restrict__ vb) {
    size_t idx = (size_t)blockIdx.x * 256 + threadIdx.x;  // over B*T*H
    int h = (int)(idx % Hc);
    size_t bt = idx / Hc;
    int t = (int)(bt % Tc);
    size_t b = bt / Tc;
    float acc = bc[h];
#pragma unroll
    for (int k = 0; k < Kc; ++k) {
        int tt = t - (Kc - 1) + k;
        if (tt >= 0)
            acc += gi[((b * Tc + tt) * (size_t)(2 * Hc)) + Hc + h] * Wc[h * Kc + k];
    }
    vf[idx] = acc;
    vb[idx] = f2bf(acc);
}

// ---------------- gates + sequential diagonal recurrence ----------------
__global__ __launch_bounds__(256)
void k_scan(const float* __restrict__ g, const float* __restrict__ vf,
            const float* __restrict__ gi, const float* __restrict__ alpha_p,
            u16* __restrict__ y) {
    int h = blockIdx.x * 256 + threadIdx.x;
    int b = blockIdx.y;
    float ap = fmaxf(alpha_p[0], 0.0f);
    float state = 0.0f;
    for (int t = 0; t < Tc; ++t) {
        size_t bt = (size_t)b * Tc + t;
        float fv   = g[bt * (size_t)(3 * Hc) + h];
        float iv   = g[bt * (size_t)(3 * Hc) + Hc + h];
        float ov   = g[bt * (size_t)(3 * Hc) + 2 * Hc + h];
        float vv   = vf[bt * (size_t)Hc + h];
        float gate = gi[bt * (size_t)(2 * Hc) + h];
        float sig_nf = 1.0f / (1.0f + expf(fv));          // sigmoid(-f)
        float alpha  = powf(sig_nf, ap);
        float xin    = vv / (1.0f + expf(-iv));           // sigmoid(i)*v
        state = alpha * state + xin;
        float so = 1.0f / (1.0f + expf(-ov));
        float gl = 0.5f * gate * (1.0f + erff(gate * 0.7071067811865475f));
        y[bt * (size_t)Hc + h] = f2bf(gl * so * state);
    }
}

// ---------------- bf16 WMMA GEMM:  C[M,N] = A[M,K] * B[N,K]^T (+bias) ----------------
// 128x128 block tile, 256 threads = 8 waves, wave tile 32(M) x 64(N) = 2x4 WMMA tiles.
// Double-buffered LDS; A staged by per-lane GLOBAL_LOAD_ASYNC_TO_LDS_B128 (ASYNCcnt),
// B (weights) staged by the Tensor Data Mover (TENSORcnt) with LDS row padding.
constexpr int TILE_K    = 32;
constexpr int LDS_PITCH = 40;               // 32 + 8 pad elems -> 80B rows (16B aligned)
constexpr unsigned BUF_BYTES = 128 * LDS_PITCH * 2;

__global__ __launch_bounds__(256)
void k_gemm_bf16(const u16* __restrict__ A, const u16* __restrict__ Bm,
                 const float* __restrict__ bias, float* __restrict__ C,
                 int M, int N, int Kd) {
    __shared__ __align__(16) u16 ldsA[2][128 * LDS_PITCH];
    __shared__ __align__(16) u16 ldsB[2][128 * LDS_PITCH];

    const int tid  = threadIdx.x;
    const int lane = tid & 31;
    const int wave = tid >> 5;
    const int wm   = wave & 3;
    const int wn   = wave >> 2;
    const int l16  = lane & 15;
    const int half = lane >> 4;
    const int mBlock = blockIdx.y * 128;
    const int nBlock = blockIdx.x * 128;

    const unsigned ldsA0 = (unsigned)(uintptr_t)&ldsA[0][0];
    const unsigned ldsB0 = (unsigned)(uintptr_t)&ldsB[0][0];

    v8f acc[2][4];
#pragma unroll
    for (int tm = 0; tm < 2; ++tm)
#pragma unroll
        for (int tn = 0; tn < 4; ++tn)
#pragma unroll
            for (int e = 0; e < 8; ++e) acc[tm][tn][e] = 0.0f;

    // per-lane async copy of the A tile: 512 16B chunks, 2 per thread
    auto stageA = [&](int buf, int ko) {
#pragma unroll
        for (int i = 0; i < 2; ++i) {
            int c = tid + i * 256;
            int row = c >> 2, kc = c & 3;
            unsigned la = ldsA0 + (unsigned)buf * BUF_BYTES +
                          (unsigned)(row * LDS_PITCH + kc * 8) * 2u;
            unsigned long long ga =
                (unsigned long long)(A + (size_t)(mBlock + row) * Kd + ko + kc * 8);
            asm volatile("global_load_async_to_lds_b128 %0, %1, off"
                         :: "v"(la), "v"(ga) : "memory");
        }
    };

    // TDM copy of the B tile: 128 rows x 32 bf16, row stride Kd elems,
    // LDS pad 16B every 64B to realize LDS_PITCH. Wave-uniform -> scalar branch,
    // so exactly one TDM issue per workgroup per K-step (EXEC is ignored by TDM).
    auto stageB = [&](int buf, int ko) {
        if (wave == 0) {
            unsigned lb = ldsB0 + (unsigned)buf * BUF_BYTES;
            unsigned long long gb =
                (unsigned long long)(Bm + (size_t)nBlock * Kd + ko);
            unsigned kd = (unsigned)Kd;
            u32x4 g0;
            g0[0] = 1u;                                   // count=1, user mode
            g0[1] = lb;                                   // lds_addr
            g0[2] = (unsigned)gb;                         // global_addr[31:0]
            g0[3] = (unsigned)((gb >> 32) & 0x01FFFFFFull) | 0x80000000u; // addr[56:32]|type=2
            u32x8 g1;
            g1[0] = (1u << 16) | (1u << 20) | (3u << 22) | (3u << 25);
                    // data_size=2B, pad_enable, pad_interval=16DW, pad_amount=4DW
            g1[1] = (kd & 0xFFFFu) << 16;                 // tensor_dim0[15:0]
            g1[2] = (kd >> 16) | (128u << 16);            // tensor_dim0[31:16], tensor_dim1=128
            g1[3] = (32u << 16);                          // tile_dim0 = 32 elems
            g1[4] = 128u;                                 // tile_dim1 = 128 rows
            g1[5] = kd;                                   // tensor_dim0_stride[31:0]
            g1[6] = 0u;
            g1[7] = 0u;
            asm volatile("tensor_load_to_lds %0, %1" :: "s"(g0), "s"(g1) : "memory");
        }
    };

    stageA(0, 0);
    stageB(0, 0);

    const int nIter = Kd / TILE_K;
    for (int it = 0; it < nIter; ++it) {
        const int cur = it & 1;
        // wait for this wave's copies into buffer `cur`
        asm volatile("s_wait_asynccnt 0x0" ::: "memory");
        if (wave == 0) __builtin_amdgcn_s_wait_tensorcnt(0);
        __syncthreads();
        // kick off copies for the next K-slab into the other buffer
        if (it + 1 < nIter) {
            stageA(cur ^ 1, (it + 1) * TILE_K);
            stageB(cur ^ 1, (it + 1) * TILE_K);
        }

        // A fragment (16-bit 16x32): lane = row M=l16; half picks K {h*8..} & {16+h*8..}
        v16bf af[2];
#pragma unroll
        for (int tm = 0; tm < 2; ++tm) {
            int row = wm * 32 + tm * 16 + l16;
            v8bf lo = *(const v8bf*)&ldsA[cur][row * LDS_PITCH + half * 8];
            v8bf hi = *(const v8bf*)&ldsA[cur][row * LDS_PITCH + 16 + half * 8];
#pragma unroll
            for (int e = 0; e < 8; ++e) { af[tm][e] = lo[e]; af[tm][e + 8] = hi[e]; }
        }
        // B fragment (16-bit 32x16): lane = col N=l16; half picks K {h*8..} & {16+h*8..}
        v16bf bfr[4];
#pragma unroll
        for (int tn = 0; tn < 4; ++tn) {
            int row = wn * 64 + tn * 16 + l16;
            v8bf lo = *(const v8bf*)&ldsB[cur][row * LDS_PITCH + half * 8];
            v8bf hi = *(const v8bf*)&ldsB[cur][row * LDS_PITCH + 16 + half * 8];
#pragma unroll
            for (int e = 0; e < 8; ++e) { bfr[tn][e] = lo[e]; bfr[tn][e + 8] = hi[e]; }
        }

#pragma unroll
        for (int tm = 0; tm < 2; ++tm)
#pragma unroll
            for (int tn = 0; tn < 4; ++tn)
                acc[tm][tn] = __builtin_amdgcn_wmma_f32_16x16x32_bf16(
                    false, af[tm], false, bfr[tn], (short)0, acc[tm][tn], false, false);
    }

    // epilogue: C/D layout -> lane half selects M rows 0-7 / 8-15, l16 = column
#pragma unroll
    for (int tm = 0; tm < 2; ++tm) {
        int rbase = mBlock + wm * 32 + tm * 16 + half * 8;
#pragma unroll
        for (int tn = 0; tn < 4; ++tn) {
            int col = nBlock + wn * 64 + tn * 16 + l16;
            float bv = bias ? bias[col] : 0.0f;
#pragma unroll
            for (int v = 0; v < 8; ++v)
                C[(size_t)(rbase + v) * N + col] = acc[tm][tn][v] + bv;
        }
    }
}

// ---------------- host orchestration ----------------
extern "C" void kernel_launch(void* const* d_in, const int* in_sizes, int n_in,
                              void* d_out, int out_size, void* d_ws, size_t ws_size,
                              hipStream_t stream) {
    const float* x       = (const float*)d_in[0];   // [B,T,D]
    const float* W_in    = (const float*)d_in[1];   // [2H,D]
    const float* W_conv  = (const float*)d_in[2];   // [H,1,K]
    const float* b_conv  = (const float*)d_in[3];   // [H]
    const float* W_gates = (const float*)d_in[4];   // [3H,H]
    const float* b_gates = (const float*)d_in[5];   // [3H]
    const float* W_out   = (const float*)d_in[6];   // [D,H]
    const float* alpha_p = (const float*)d_in[7];   // [1]
    float* out = (float*)d_out;                     // [B,T,D]

    char* ws = (char*)d_ws;
    size_t off = 0;
    auto take = [&](size_t bytes) { char* p = ws + off; off += (bytes + 255) & ~size_t(255); return p; };
    u16*   x_bf   = (u16*)  take((size_t)MT * Dc * 2);
    u16*   Win_bf = (u16*)  take((size_t)2 * Hc * Dc * 2);
    u16*   Wg_bf  = (u16*)  take((size_t)3 * Hc * Hc * 2);
    u16*   Wo_bf  = (u16*)  take((size_t)Dc * Hc * 2);
    float* gi     = (float*)take((size_t)MT * 2 * Hc * 4);
    float* vf     = (float*)take((size_t)MT * Hc * 4);
    u16*   v_bf   = (u16*)  take((size_t)MT * Hc * 2);
    float* g      = (float*)take((size_t)MT * 3 * Hc * 4);
    u16*   y_bf   = (u16*)  take((size_t)MT * Hc * 2);
    (void)ws_size;

    k_cvt_bf16<<<2048, 256, 0, stream>>>(x,       x_bf,   (long)MT * Dc);
    k_cvt_bf16<<<2048, 256, 0, stream>>>(W_in,    Win_bf, (long)2 * Hc * Dc);
    k_cvt_bf16<<<2048, 256, 0, stream>>>(W_gates, Wg_bf,  (long)3 * Hc * Hc);
    k_cvt_bf16<<<2048, 256, 0, stream>>>(W_out,   Wo_bf,  (long)Dc * Hc);

    k_gemm_bf16<<<dim3((2 * Hc) / 128, MT / 128), 256, 0, stream>>>(
        x_bf, Win_bf, nullptr, gi, MT, 2 * Hc, Dc);

    k_conv<<<(size_t)MT * Hc / 256, 256, 0, stream>>>(gi, W_conv, b_conv, vf, v_bf);

    k_gemm_bf16<<<dim3((3 * Hc) / 128, MT / 128), 256, 0, stream>>>(
        v_bf, Wg_bf, b_gates, g, MT, 3 * Hc, Hc);

    k_scan<<<dim3(Hc / 256, Bc), 256, 0, stream>>>(g, vf, gi, alpha_p, y_bf);

    k_gemm_bf16<<<dim3(Dc / 128, MT / 128), 256, 0, stream>>>(
        y_bf, Wo_bf, nullptr, out, MT, Dc, Hc);
}
